// PairwiseGPMapLayer_1108101563157
// MI455X (gfx1250) — matrix operand order repaired and checked by reading, not compiled, not run on hardware
//
#include <hip/hip_runtime.h>

#define LPOS  256
#define CCH   20
#define LCDIM 5120          // LPOS * CCH
#define BATCH 1024

typedef __bf16 v16bf __attribute__((ext_vector_type(16)));
typedef __bf16 v8bf  __attribute__((ext_vector_type(8)));
typedef __bf16 v4bf  __attribute__((ext_vector_type(4)));
typedef float  v8f   __attribute__((ext_vector_type(8)));

// ---------------------------------------------------------------------------
// Prep 1: W[n*LC + k] = bf16( theta_lclc[n*LC + k] * (l(k) > l(n)) )
// Using x^T M x == x^T M^T x, this elementwise masked copy (no transpose!)
// is directly usable as the GEMM B matrix with contiguous per-column K loads.
// ---------------------------------------------------------------------------
__global__ void prep_theta(const float4* __restrict__ th, v4bf* __restrict__ W) {
    unsigned idx  = blockIdx.x * 256u + threadIdx.x;   // group of 4 elements
    unsigned base = idx * 4u;
    unsigned n = base / LCDIM;
    unsigned k = base % LCDIM;                          // LC % 4 == 0 -> same n for all 4
    int ln = (int)(n / CCH);
    float4 v = th[idx];
    v4bf o;
    o[0] = (__bf16)((int)((k + 0u) / CCH) > ln ? v.x : 0.0f);
    o[1] = (__bf16)((int)((k + 1u) / CCH) > ln ? v.y : 0.0f);
    o[2] = (__bf16)((int)((k + 2u) / CCH) > ln ? v.z : 0.0f);
    o[3] = (__bf16)((int)((k + 3u) / CCH) > ln ? v.w : 0.0f);
    W[idx] = o;
}

// Prep 2: bf16 copy of x (A matrix source)
__global__ void prep_x(const float4* __restrict__ x, v4bf* __restrict__ Xb) {
    unsigned idx = blockIdx.x * 256u + threadIdx.x;
    float4 v = x[idx];
    v4bf o;
    o[0] = (__bf16)v.x; o[1] = (__bf16)v.y; o[2] = (__bf16)v.z; o[3] = (__bf16)v.w;
    Xb[idx] = o;
}

__global__ void zero_quad(float* __restrict__ quad) {
    unsigned i = blockIdx.x * 256u + threadIdx.x;
    if (i < BATCH) quad[i] = 0.0f;
}

static __device__ __forceinline__ v16bf cat16(v8bf lo, v8bf hi) {
    return __builtin_shufflevector(lo, hi, 0,1,2,3,4,5,6,7,8,9,10,11,12,13,14,15);
}

// ---------------------------------------------------------------------------
// GEMM + fused quad reduction.
// Block = 4 waves, 128 threads. All waves share ONE 64-column group (B
// fragments identical across waves -> WGP$ dedup); each wave owns a distinct
// 32-row batch tile -> wave tile 32x64 = 2 (M) x 4 (N) accumulators.
// Per K-step (K += 32): 4 A-loads + 8 B-loads (b128) feed 8 WMMAs
// -> ~43 FLOP per unique L2 byte per block. K loop unrolled x2 so the
// scheduler can keep next-step loads in flight across the WMMA chain.
//
// A-fragment layout (16-bit A 16x32, ISA 7.12.2):
//   lane<16 : row=lane,    K = {k0..k0+7} u {k0+16..k0+23}
//   lane>=16: row=lane-16, K = {k0+8..k0+15} u {k0+24..k0+31}
// B-fragment (16-bit B 32x16): same K split as A: lane<16 col=lane,
//   K = {k0..k0+7} u {k0+16..k0+23}; lanes 16-31: {k0+8..k0+15} u {k0+24..k0+31}.
// ---------------------------------------------------------------------------
__global__ void __launch_bounds__(128) gemm_quad(
        const __bf16* __restrict__ Xb,     // [BATCH, LC] bf16
        const __bf16* __restrict__ W,      // [LC, LC] bf16 (masked)
        const float*  __restrict__ x,      // [BATCH, LC] fp32 (epilogue product)
        float*        __restrict__ quad)   // [BATCH] fp32 accumulator
{
    const int lane    = threadIdx.x & 31;
    const int wave    = threadIdx.x >> 5;
    const int half    = lane >> 4;        // 0: lanes 0-15, 1: lanes 16-31
    const int l15     = lane & 15;
    const int m0      = (blockIdx.y * 4 + wave) * 32;  // 32 batch rows per wave
    const int colbase = blockIdx.x * 64;               // 64 cols per block (shared)

    v8f acc[2][4] = {};

    const __bf16* arow0 = Xb + (size_t)(m0 +  0 + l15) * LCDIM + half * 8;
    const __bf16* arow1 = Xb + (size_t)(m0 + 16 + l15) * LCDIM + half * 8;
    const __bf16* bp[4];
    #pragma unroll
    for (int nt = 0; nt < 4; ++nt)
        bp[nt] = W + (size_t)(colbase + nt * 16 + l15) * LCDIM + half * 8;

    #pragma unroll 2
    for (int k0 = 0; k0 < LCDIM; k0 += 32) {
        // pull next K-step of B into cache (global_prefetch_b8)
        __builtin_prefetch(bp[0] + k0 + 64, 0, 1);
        __builtin_prefetch(bp[2] + k0 + 64, 0, 1);

        v16bf a0 = cat16(*(const v8bf*)(arow0 + k0), *(const v8bf*)(arow0 + k0 + 16));
        v16bf a1 = cat16(*(const v8bf*)(arow1 + k0), *(const v8bf*)(arow1 + k0 + 16));

        v16bf f[4];
        #pragma unroll
        for (int nt = 0; nt < 4; ++nt)
            f[nt] = cat16(*(const v8bf*)(bp[nt] + k0), *(const v8bf*)(bp[nt] + k0 + 16));

        #pragma unroll
        for (int nt = 0; nt < 4; ++nt) {
            acc[0][nt] = __builtin_amdgcn_wmma_f32_16x16x32_bf16(
                false, a0, false, f[nt], (short)0, acc[0][nt], false, false);
            acc[1][nt] = __builtin_amdgcn_wmma_f32_16x16x32_bf16(
                false, a1, false, f[nt], (short)0, acc[1][nt], false, false);
        }
    }

    // Epilogue: quad[m] += sum_n xm[m,n] * x[m,n] over this block's 64 columns.
    // C/D layout: VGPR r, lanes 0-15 -> (M=r, N=lane); lanes 16-31 -> (M=r+8, N=lane-16).
    #pragma unroll
    for (int mt = 0; mt < 2; ++mt) {
        #pragma unroll
        for (int r = 0; r < 8; ++r) {
            const int m = m0 + mt * 16 + half * 8 + r;
            const float* xm = x + (size_t)m * LCDIM + colbase + l15;
            float v = acc[mt][0][r] * xm[0]  + acc[mt][1][r] * xm[16]
                    + acc[mt][2][r] * xm[32] + acc[mt][3][r] * xm[48];
            // reduce across the 16 lanes of this half (wave32 shuffles)
            #pragma unroll
            for (int off = 1; off < 16; off <<= 1)
                v += __shfl_xor(v, off, 32);
            if (l15 == 0) atomicAdd(&quad[m], v);
        }
    }
}

// ---------------------------------------------------------------------------
// Finalize: out[b] = theta0 + dot(x[b], theta_lc) + quad[b]   (all fp32)
// ---------------------------------------------------------------------------
__global__ void __launch_bounds__(256) finalize(
        const float* __restrict__ x, const float* __restrict__ theta0,
        const float* __restrict__ theta_lc, const float* __restrict__ quad,
        float* __restrict__ out)
{
    __shared__ float red[256];
    const int b = blockIdx.x;
    const float* xb = x + (size_t)b * LCDIM;
    float s = 0.0f;
    for (int i = threadIdx.x; i < LCDIM; i += 256) s += xb[i] * theta_lc[i];
    red[threadIdx.x] = s;
    __syncthreads();
    for (int o = 128; o > 0; o >>= 1) {
        if (threadIdx.x < o) red[threadIdx.x] += red[threadIdx.x + o];
        __syncthreads();
    }
    if (threadIdx.x == 0) out[b] = theta0[0] + red[0] + quad[b];
}

extern "C" void kernel_launch(void* const* d_in, const int* in_sizes, int n_in,
                              void* d_out, int out_size, void* d_ws, size_t ws_size,
                              hipStream_t stream) {
    const float* x          = (const float*)d_in[0];   // [1024, 5120]
    const float* theta_0    = (const float*)d_in[1];   // [1]
    const float* theta_lc   = (const float*)d_in[2];   // [5120]
    const float* theta_lclc = (const float*)d_in[3];   // [5120*5120]

    char* ws = (char*)d_ws;
    __bf16* W    = (__bf16*)ws;                                              // 52,428,800 B
    __bf16* Xb   = (__bf16*)(ws + (size_t)LCDIM * LCDIM * 2);                // 10,485,760 B
    float*  quad = (float*)(ws + (size_t)LCDIM * LCDIM * 2
                               + (size_t)BATCH * LCDIM * 2);                 //      4,096 B

    // 1) masked bf16 Theta (coalesced, no transpose needed by x^T M x symmetry)
    prep_theta<<<(LCDIM * LCDIM) / (4 * 256), 256, 0, stream>>>(
        (const float4*)theta_lclc, (v4bf*)W);
    // 2) bf16 x
    prep_x<<<(BATCH * LCDIM) / (4 * 256), 256, 0, stream>>>(
        (const float4*)x, (v4bf*)Xb);
    // 3) clear quad accumulator (must happen every call)
    zero_quad<<<(BATCH + 255) / 256, 256, 0, stream>>>(quad);
    // 4) WMMA GEMM + fused quad reduction:
    //    grid (5120/64 col groups, 1024/(4*32) batch groups), 4 waves/block
    dim3 g(LCDIM / 64, BATCH / 128);   // (80, 8)
    gemm_quad<<<g, 128, 0, stream>>>(Xb, W, x, quad);
    // 5) linear term + assemble output
    finalize<<<BATCH, 256, 0, stream>>>(x, theta_0, theta_lc, quad, (float*)d_out);
}
